// ResidualUpsampleBKPConv_2370821947673
// MI455X (gfx1250) — matrix operand order repaired and checked by reading, not compiled, not run on hardware
//
#include <hip/hip_runtime.h>
#include <hip/hip_bf16.h>
#include <math.h>

typedef __attribute__((ext_vector_type(16))) _Float16 v16h;
typedef __attribute__((ext_vector_type(8)))  float    v8f;

#define N_PTS   16384
#define NS_PTS  65536
#define E_EDGES 1048576
#define C_IN    256
#define C_MID   64
#define C_OUT   128
#define KPTS    16
#define C_SKIP  128
#define RADIUS  0.2f

// ---------------- workspace layout (bytes) ----------------
#define WS_AGG       0
#define WS_SHORTACC  16777216
#define WS_BNSUM     50331648
#define WS_BNSQ      50332160
#define WS_XSIDE     50332672
#define WS_XSHORT    52429824
#define WS_BFRAG     60818432

// ---------------------------------------------------------------------------
// K0: zero accumulators (agg + short_acc + bn sums are contiguous floats)
// ---------------------------------------------------------------------------
__global__ void zero_ws(float* p, int n) {
  for (int i = blockIdx.x * blockDim.x + threadIdx.x; i < n; i += gridDim.x * blockDim.x)
    p[i] = 0.0f;
}

// ---------------------------------------------------------------------------
// K1: x_side = (x @ pre_W + pre_b) -> f16 table  [N,64]
// ---------------------------------------------------------------------------
__global__ void pre_gemm(const float* __restrict__ x, const float* __restrict__ W,
                         const float* __restrict__ b, _Float16* __restrict__ out) {
  __shared__ float xr[C_IN];
  const int n = blockIdx.x;
  for (int i = threadIdx.x; i < C_IN; i += 64) xr[i] = x[(size_t)n * C_IN + i];
  __syncthreads();
  const int c = threadIdx.x;            // 64 threads
  float acc = b[c];
  #pragma unroll 8
  for (int i = 0; i < C_IN; ++i) acc += xr[i] * W[i * C_MID + c];
  out[(size_t)n * C_MID + c] = (_Float16)acc;
}

// ---------------------------------------------------------------------------
// K2: x_short = x @ short_W + short_b   [N,128] fp32
// ---------------------------------------------------------------------------
__global__ void short_gemm(const float* __restrict__ x, const float* __restrict__ W,
                           const float* __restrict__ b, float* __restrict__ out) {
  __shared__ float xr[C_IN];
  const int n = blockIdx.x;
  for (int i = threadIdx.x; i < C_IN; i += 128) xr[i] = x[(size_t)n * C_IN + i];
  __syncthreads();
  const int c = threadIdx.x;            // 128 threads
  float acc = b[c];
  #pragma unroll 8
  for (int i = 0; i < C_IN; ++i) acc += xr[i] * W[i * C_OUT + c];
  out[(size_t)n * C_OUT + c] = acc;
}

// ---------------------------------------------------------------------------
// K3: pack kernel_weight[k][64][64] into f16 WMMA B-fragments.
// Fragment f = (k*2 + kt)*4 + nt; element index = lane*16 + slot.
// B 32x16 layout: lane holds column N=lane&15, K = slot + 16*(lane>>4),
// global inner index in = kt*32 + K.
// ---------------------------------------------------------------------------
__global__ void pack_b(const float* __restrict__ kw, _Float16* __restrict__ bf) {
  const int t    = blockIdx.x * 256 + threadIdx.x;   // 65536 total
  const int slot = t & 15;
  const int lane = (t >> 4) & 31;
  const int nt   = (t >> 9) & 3;
  const int kt   = (t >> 11) & 1;
  const int k    = t >> 12;
  const int hi   = lane >> 4;
  const int in   = kt * 32 + slot + 16 * hi;
  const int oc   = nt * 16 + (lane & 15);
  bf[t] = (_Float16)kw[((size_t)k * 64 + in) * 64 + oc];
}

// ---------------------------------------------------------------------------
// K4: KPConv edge messages with WMMA + atomic scatter into agg.
// Block = 256 threads (8 waves); each wave processes 4 tiles of 16 edges.
// B fragments (128KB) staged in LDS once per block; the k-loop is kept
// rolled (#pragma unroll 1) so B operands stream LDS -> WMMA without
// register spills.
// ---------------------------------------------------------------------------
#define MSG_WAVES 8
#define MSG_TILES 4

__global__ void __launch_bounds__(256)
msg_kernel(const _Float16* __restrict__ xside,
           const _Float16* __restrict__ bfrag_g,
           const int* __restrict__ row, const int* __restrict__ col,
           const float* __restrict__ pos,
           const float* __restrict__ pos_skip,
           const float* __restrict__ kp,
           float* __restrict__ agg) {
  extern __shared__ _Float16 sB[];                    // 16*4096 halfs = 128KB
  for (int i = threadIdx.x; i < KPTS * 4096; i += blockDim.x) sB[i] = bfrag_g[i];
  __syncthreads();

  const int wave = threadIdx.x >> 5;
  const int lane = threadIdx.x & 31;
  const int hi   = lane >> 4;
  const int m    = lane & 15;

  #pragma unroll 1
  for (int t = 0; t < MSG_TILES; ++t) {
    const int tile = (blockIdx.x * MSG_WAVES + wave) * MSG_TILES + t;
    const int base = tile * 16;
    const int e    = base + m;                        // lanes 16-31 mirror 0-15
    const int ri   = row[e];
    const int ci   = col[e];

    const float rx = pos[ci * 3 + 0] - pos_skip[ri * 3 + 0];
    const float ry = pos[ci * 3 + 1] - pos_skip[ri * 3 + 1];
    const float rz = pos[ci * 3 + 2] - pos_skip[ri * 3 + 2];

    // A fragments (16x32 f16): lane holds row m, K = 16*hi + s within each 32-wide chunk
    const _Float16* xr = xside + (size_t)ci * C_MID;
    v16h a0, a1;
    ((uint4*)&a0)[0] = *(const uint4*)(xr +  0 + 16 * hi);
    ((uint4*)&a0)[1] = *(const uint4*)(xr +  8 + 16 * hi);
    ((uint4*)&a1)[0] = *(const uint4*)(xr + 32 + 16 * hi);
    ((uint4*)&a1)[1] = *(const uint4*)(xr + 40 + 16 * hi);

    v8f acc[4];
    #pragma unroll
    for (int nt = 0; nt < 4; ++nt) acc[nt] = (v8f){};

    #pragma unroll 1
    for (int k = 0; k < KPTS; ++k) {
      const float dx = rx - kp[k * 3 + 0];
      const float dy = ry - kp[k * 3 + 1];
      const float dz = rz - kp[k * 3 + 2];
      const float d  = sqrtf(dx * dx + dy * dy + dz * dz);
      const float infl = fmaxf(1.0f - d * (1.0f / RADIUS), 0.0f);
      const _Float16 s = (_Float16)infl;
      const v16h as0 = a0 * s;
      const v16h as1 = a1 * s;
      const _Float16* bk = sB + k * 4096;
      #pragma unroll
      for (int nt = 0; nt < 4; ++nt) {
        const v16h b0 = *(const v16h*)(bk + nt * 512 + lane * 16);
        acc[nt] = __builtin_amdgcn_wmma_f32_16x16x32_f16(
            false, as0, false, b0, (short)0, acc[nt], false, false);
        const v16h b1 = *(const v16h*)(bk + 2048 + nt * 512 + lane * 16);
        acc[nt] = __builtin_amdgcn_wmma_f32_16x16x32_f16(
            false, as1, false, b1, (short)0, acc[nt], false, false);
      }
    }

    // scatter: C/D layout -> VGPR r, lane: row M = r + 8*hi, col N = nt*16 + m
    #pragma unroll
    for (int nt = 0; nt < 4; ++nt) {
      #pragma unroll
      for (int r = 0; r < 8; ++r) {
        const int trow = __shfl(ri, r + 8 * hi, 32);
        atomicAdd(agg + (size_t)trow * C_MID + nt * 16 + m, acc[nt][r]);
      }
    }
  }
}

// ---------------------------------------------------------------------------
// K5: shortcut scatter: short_acc[row[e]] += x_short[col[e]]
// ---------------------------------------------------------------------------
__global__ void short_scatter(const float* __restrict__ xs,
                              const int* __restrict__ row, const int* __restrict__ col,
                              float* __restrict__ acc) {
  const long idx = (long)blockIdx.x * 256 + threadIdx.x;   // E*32
  const int e  = (int)(idx >> 5);
  const int c4 = (int)(idx & 31) << 2;
  const int r = row[e], c = col[e];
  const float4 v = *(const float4*)(xs + (size_t)c * C_OUT + c4);
  float* p = acc + (size_t)r * C_OUT + c4;
  atomicAdd(p + 0, v.x);
  atomicAdd(p + 1, v.y);
  atomicAdd(p + 2, v.z);
  atomicAdd(p + 3, v.w);
}

// ---------------------------------------------------------------------------
// K6: out = agg@post_W + post_b + short_acc; h = [out,x_skip]@mlp_W + mlp_b
// All weights (160KB) staged in LDS; 64 rows per block; channel sums for BN.
// ---------------------------------------------------------------------------
#define MLP_ROWS 64

__global__ void combine_mlp(const float* __restrict__ agg,
                            const float* __restrict__ shortacc,
                            const float* __restrict__ xskip,
                            const float* __restrict__ postW, const float* __restrict__ postb,
                            const float* __restrict__ mlpW,  const float* __restrict__ mlpb,
                            float* __restrict__ hout,
                            float* __restrict__ bnsum, float* __restrict__ bnsq) {
  extern __shared__ float sm[];
  float* wpost = sm;                 // 64*128   = 8192
  float* wmlp  = sm + 8192;          // 256*128  = 32768
  float* vrow  = sm + 8192 + 32768;  // 128
  float* arow  = vrow + 128;         // 64
  float* srow  = arow + 64;          // 128
  const int tid = threadIdx.x;       // 128 threads

  for (int i = tid; i < 8192;  i += 128) wpost[i] = postW[i];
  for (int i = tid; i < 32768; i += 128) wmlp[i]  = mlpW[i];
  __syncthreads();

  const float pb = postb[tid];
  const float mb = mlpb[tid];

  #pragma unroll 1
  for (int rr = 0; rr < MLP_ROWS; ++rr) {
    const int i = blockIdx.x * MLP_ROWS + rr;
    if (tid < 64) arow[tid] = agg[(size_t)i * C_MID + tid];
    srow[tid] = xskip[(size_t)i * C_SKIP + tid];
    __syncthreads();

    float s = pb + shortacc[(size_t)i * C_OUT + tid];
    #pragma unroll 8
    for (int j = 0; j < 64; ++j) s += arow[j] * wpost[j * 128 + tid];
    vrow[tid] = s;
    __syncthreads();

    float h = mb;
    #pragma unroll 8
    for (int j = 0; j < 128; ++j) h += vrow[j] * wmlp[j * 128 + tid];
    #pragma unroll 8
    for (int j = 0; j < 128; ++j) h += srow[j] * wmlp[(128 + j) * 128 + tid];

    hout[(size_t)i * C_OUT + tid] = h;
    atomicAdd(&bnsum[tid], h);
    atomicAdd(&bnsq[tid], h * h);
    __syncthreads();
  }
}

// ---------------------------------------------------------------------------
// K7: BatchNorm (training batch stats) + LeakyReLU(0.2), in place on d_out
// ---------------------------------------------------------------------------
__global__ void bn_final(float* __restrict__ h,
                         const float* __restrict__ bnsum, const float* __restrict__ bnsq,
                         const float* __restrict__ gamma, const float* __restrict__ beta) {
  const int idx = blockIdx.x * 256 + threadIdx.x;          // NS*128
  const int c = idx & (C_OUT - 1);
  const float inv_n = 1.0f / (float)NS_PTS;
  const float mu  = bnsum[c] * inv_n;
  const float var = bnsq[c] * inv_n - mu * mu;
  float v = (h[idx] - mu) * rsqrtf(var + 1e-5f) * gamma[c] + beta[c];
  h[idx] = (v >= 0.0f) ? v : 0.2f * v;
}

// ---------------------------------------------------------------------------
extern "C" void kernel_launch(void* const* d_in, const int* in_sizes, int n_in,
                              void* d_out, int out_size, void* d_ws, size_t ws_size,
                              hipStream_t stream) {
  const float* x        = (const float*)d_in[0];
  const float* pos      = (const float*)d_in[1];
  const float* pos_skip = (const float*)d_in[2];
  const float* x_skip   = (const float*)d_in[3];
  const float* pre_W    = (const float*)d_in[4];
  const float* pre_b    = (const float*)d_in[5];
  const float* kpts     = (const float*)d_in[6];
  const float* kweight  = (const float*)d_in[7];
  const float* post_W   = (const float*)d_in[8];
  const float* post_b   = (const float*)d_in[9];
  const float* short_W  = (const float*)d_in[10];
  const float* short_b  = (const float*)d_in[11];
  const float* mlp_W    = (const float*)d_in[12];
  const float* mlp_b    = (const float*)d_in[13];
  const float* bn_gamma = (const float*)d_in[14];
  const float* bn_beta  = (const float*)d_in[15];
  const int*   row      = (const int*)d_in[16];
  const int*   col      = (const int*)d_in[17];

  char* ws = (char*)d_ws;
  float*     agg     = (float*)(ws + WS_AGG);
  float*     sht     = (float*)(ws + WS_SHORTACC);
  float*     bnsum   = (float*)(ws + WS_BNSUM);
  float*     bnsq    = (float*)(ws + WS_BNSQ);
  _Float16*  xside   = (_Float16*)(ws + WS_XSIDE);
  float*     xshort  = (float*)(ws + WS_XSHORT);
  _Float16*  bfrag   = (_Float16*)(ws + WS_BFRAG);
  float*     hout    = (float*)d_out;

  // K0: zero agg + short_acc + bn sums (contiguous: NS*192 + 256 floats)
  zero_ws<<<4096, 256, 0, stream>>>((float*)ws, NS_PTS * 192 + 256);

  // K1/K2: input-side GEMMs
  pre_gemm  <<<N_PTS, 64,  0, stream>>>(x, pre_W,  pre_b,  xside);
  short_gemm<<<N_PTS, 128, 0, stream>>>(x, short_W, short_b, xshort);

  // K3: pack kernel weights into WMMA B fragments
  pack_b<<<256, 256, 0, stream>>>(kweight, bfrag);

  // K4: KPConv messages (WMMA) + scatter into agg
  msg_kernel<<<E_EDGES / (16 * MSG_WAVES * MSG_TILES), 256,
               KPTS * 4096 * sizeof(_Float16), stream>>>(
      xside, bfrag, row, col, pos, pos_skip, kpts, agg);

  // K5: shortcut scatter
  short_scatter<<<(E_EDGES * 32) / 256, 256, 0, stream>>>(xshort, row, col, sht);

  // K6: post GEMM + residual + MLP (+ BN partial sums)
  combine_mlp<<<NS_PTS / MLP_ROWS, 128, (8192 + 32768 + 128 + 64 + 128) * sizeof(float),
                stream>>>(agg, sht, x_skip, post_W, post_b, mlp_W, mlp_b,
                          hout, bnsum, bnsq);

  // K7: BatchNorm + LeakyReLU
  bn_final<<<(NS_PTS * C_OUT) / 256, 256, 0, stream>>>(hout, bnsum, bnsq, bn_gamma, bn_beta);
}
